// SE3GroupConvLayer_47571057770952
// MI455X (gfx1250) — matrix operand order repaired and checked by reading, not compiled, not run on hardware
//
#include <hip/hip_runtime.h>
#include <math.h>

typedef __attribute__((ext_vector_type(16))) _Float16 v16h;
typedef __attribute__((ext_vector_type(8)))  _Float16 v8h;
typedef __attribute__((ext_vector_type(8)))  float    v8f;

#define CCH 16
#define FDIM 64

// CDNA5 16-bit A-fragment (16x32 MxK, wave32): for lane L (M=L%16, grp=L/16),
// element h maps to K = 8*grp + h (h<8) and K = 16 + 8*grp + (h-8) (h>=8).
// => two contiguous 8-half (16B) runs of a row-major tile: ds_load_b128 x2.
union afrag_u { v16h v; v8h h[2]; };

__device__ __forceinline__ float frelu(float x) { return x > 0.f ? x : 0.f; }

// ============ 1. pose prep: Rodrigues + Wigner-D1 (l=1 permutation) ============
__global__ void prep_poses_kernel(const float* __restrict__ alg,
                                  float* __restrict__ d1inv,
                                  float* __restrict__ d1fwd, int P) {
  int p = blockIdx.x * blockDim.x + threadIdx.x;
  if (p >= P) return;
  float wx = fminf(fmaxf(alg[p * 6 + 3], -10.f), 10.f);
  float wy = fminf(fmaxf(alg[p * 6 + 4], -10.f), 10.f);
  float wz = fminf(fmaxf(alg[p * 6 + 5], -10.f), 10.f);
  float th = sqrtf(wx * wx + wy * wy + wz * wz);
  float R[3][3];
  if (th < 1e-8f) {
    R[0][0] = 1.f; R[0][1] = 0.f; R[0][2] = 0.f;
    R[1][0] = 0.f; R[1][1] = 1.f; R[1][2] = 0.f;
    R[2][0] = 0.f; R[2][1] = 0.f; R[2][2] = 1.f;
  } else {
    float kx = wx / th, ky = wy / th, kz = wz / th;
    float st = sinf(th), ct = cosf(th), vt = 1.f - ct;
    R[0][0] = ct + kx * kx * vt;      R[0][1] = kx * ky * vt - kz * st; R[0][2] = kx * kz * vt + ky * st;
    R[1][0] = ky * kx * vt + kz * st; R[1][1] = ct + ky * ky * vt;      R[1][2] = ky * kz * vt - kx * st;
    R[2][0] = kz * kx * vt - ky * st; R[2][1] = kz * ky * vt + kx * st; R[2][2] = ct + kz * kz * vt;
  }
  const int pi[3] = {1, 2, 0};
  for (int i = 0; i < 3; ++i)
    for (int j = 0; j < 3; ++j) {
      d1fwd[p * 9 + i * 3 + j] = R[pi[i]][pi[j]];   // D1 of R_fwd
      d1inv[p * 9 + i * 3 + j] = R[pi[j]][pi[i]];   // D1 of R^T
    }
}

// ===== 2. weight conversion to f16, TRANSPOSED + K-padded-to-32 for B frags =====
// layout: [0,1024)   W2t[n=16][k=64]
//         [1024,3072) Wpt[o=64][k=32]  (k>=16 zero)
//         [3072,3584) Wst[o=16][k=32]  (k>=16 zero)
//         [3584,4096) Wvt[o=16][k=32]  (k>=16 zero)
__global__ void convert_weights_kernel(const float* __restrict__ W2,
                                       const float* __restrict__ Wp,
                                       const float* __restrict__ Ws,
                                       const float* __restrict__ Wv,
                                       _Float16* __restrict__ wh) {
  int t = blockIdx.x * blockDim.x + threadIdx.x;
  if (t < 1024) {
    int n = t >> 6, k = t & 63;
    wh[t] = (_Float16)W2[k * 16 + n];
  } else if (t < 3072) {
    int idx = t - 1024; int o = idx >> 5, k = idx & 31;
    wh[t] = (_Float16)((k < 16) ? Wp[k * 64 + o] : 0.f);
  } else if (t < 3584) {
    int idx = t - 3072; int o = idx >> 5, k = idx & 31;
    wh[t] = (_Float16)((k < 16) ? Ws[k * 16 + o] : 0.f);
  } else if (t < 4096) {
    int idx = t - 3584; int o = idx >> 5, k = idx & 31;
    wh[t] = (_Float16)((k < 16) ? Wv[k * 16 + o] : 0.f);
  }
}

// ============ 3. align features: s passthrough, v = -D1inv v (NaN guarded) ======
__global__ void align_feats_kernel(const float* __restrict__ x,
                                   const int* __restrict__ batch,
                                   const float* __restrict__ d1inv,
                                   float* __restrict__ feats, int N_, int M_) {
  long long t = (long long)blockIdx.x * blockDim.x + threadIdx.x;
  long long total = (long long)N_ * M_ * CCH;
  if (t >= total) return;
  int c = (int)(t % CCH);
  long long nm = t / CCH;
  int i = (int)(nm % N_);
  int m = (int)(nm / N_);
  int pose = batch[i] * M_ + m;
  const float* D = d1inv + pose * 9;
  const float* xb = x + (long long)i * FDIM;
  float v0 = xb[16 + 3 * c], v1 = xb[16 + 3 * c + 1], v2 = xb[16 + 3 * c + 2];
  float o0 = -(D[0] * v0 + D[1] * v1 + D[2] * v2);
  float o1 = -(D[3] * v0 + D[4] * v1 + D[5] * v2);
  float o2 = -(D[6] * v0 + D[7] * v1 + D[8] * v2);
  float* fb = feats + nm * FDIM;
  fb[c] = xb[c];
  fb[16 + 3 * c]     = (o0 != o0) ? v0 : o0;
  fb[16 + 3 * c + 1] = (o1 != o1) ? v1 : o1;
  fb[16 + 3 * c + 2] = (o2 != o2) ? v2 : o2;
}

// ============ 4. edge geometry (shared across all M copies) ============
__global__ void edge_geom_kernel(const float* __restrict__ pos,
                                 const int* __restrict__ edge,
                                 float* __restrict__ elen,
                                 float* __restrict__ sh, int E_) {
  int e = blockIdx.x * blockDim.x + threadIdx.x;
  if (e >= E_) return;
  int r = edge[e], c = edge[E_ + e];
  float ex = pos[3 * r]     - pos[3 * c];
  float ey = pos[3 * r + 1] - pos[3 * c + 1];
  float ez = pos[3 * r + 2] - pos[3 * c + 2];
  float l = sqrtf(ex * ex + ey * ey + ez * ez);
  elen[e] = l;
  bool valid = l > 1e-8f;
  float inv = valid ? 1.f / l : 1.f;
  const float S3 = 1.7320508075688772f;
  sh[4 * e + 0] = valid ? 1.f : 0.f;
  sh[4 * e + 1] = valid ? S3 * ey * inv : 0.f;  // sqrt3*u_y
  sh[4 * e + 2] = valid ? S3 * ez * inv : 0.f;  // sqrt3*u_z
  sh[4 * e + 3] = valid ? S3 * ex * inv : 0.f;  // sqrt3*u_x
}

// ============ 5. edge MLP via WMMA: w = (relu(elen*W1+b1)@W2 + b2)@Wp + bp ======
__global__ void edge_mlp_kernel(const float* __restrict__ elen,
                                const float* __restrict__ W1,
                                const float* __restrict__ b1,
                                const _Float16* __restrict__ W2t,
                                const float* __restrict__ b2,
                                const _Float16* __restrict__ Wpt,
                                const float* __restrict__ bp,
                                _Float16* __restrict__ wbuf, int E_) {
  __shared__ _Float16 lds_act[8][16 * 64];
  __shared__ _Float16 lds_remb[8][16 * 32];   // K padded to 32 (zeros)
  int wave = threadIdx.x >> 5;
  int lane = threadIdx.x & 31;
  long long tile = (long long)blockIdx.x * 8 + wave;
  long long e0 = tile * 16;
  if (e0 < E_) {
    for (int idx = lane; idx < 16 * 64; idx += 32) {
      int m = idx >> 6, k = idx & 63;
      long long e = e0 + m;
      float el = (e < E_) ? elen[e] : 0.f;
      lds_act[wave][idx] = (_Float16)frelu(el * W1[k] + b1[k]);
    }
  }
  asm volatile("s_wait_dscnt 0" ::: "memory");
  __builtin_amdgcn_wave_barrier();
  if (e0 >= E_) return;
  int grp = lane >> 4;
  int mrow = lane & 15;
  int ncol = lane & 15;
  // GEMM1: remb(16x16,f32) = act(16x64) @ W2(64x16), two K=32 steps.
  v8f c1 = {};
#pragma unroll
  for (int kb = 0; kb < 2; ++kb) {
    afrag_u au;
    const _Float16* ab = &lds_act[wave][mrow * 64 + kb * 32 + 16 * grp];
    au.h[0] = *(const v8h*)(ab);        // K = kb*32 + 16g + 0..7
    au.h[1] = *(const v8h*)(ab + 8);    // K = kb*32 + 16g + 8..15
    v16h b = *(const v16h*)(W2t + ncol * 64 + kb * 32 + grp * 16);
    c1 = __builtin_amdgcn_wmma_f32_16x16x32_f16(false, au.v, false, b, (short)0, c1,
                                                false, false);
  }
  float bb2 = b2[ncol];
#pragma unroll
  for (int r = 0; r < 8; ++r) {
    int m = r + 8 * grp;
    lds_remb[wave][m * 32 + ncol]      = (_Float16)(c1[r] + bb2);
    lds_remb[wave][m * 32 + 16 + ncol] = (_Float16)0.f;   // K pad
  }
  asm volatile("s_wait_dscnt 0" ::: "memory");
  __builtin_amdgcn_wave_barrier();
  // GEMM2: w(16x64) = remb(16x16) @ Wp(16x64), K padded 16->32, N tiled 4x16.
  afrag_u a2;
  {
    const _Float16* rb = &lds_remb[wave][mrow * 32 + 16 * grp];
    a2.h[0] = *(const v8h*)(rb);
    a2.h[1] = *(const v8h*)(rb + 8);
  }
  for (int nc = 0; nc < 4; ++nc) {
    v16h b = *(const v16h*)(Wpt + (nc * 16 + ncol) * 32 + grp * 16);
    v8f z = {};
    v8f d = __builtin_amdgcn_wmma_f32_16x16x32_f16(false, a2.v, false, b, (short)0, z,
                                                   false, false);
    float bbp = bp[nc * 16 + ncol];
#pragma unroll
    for (int r = 0; r < 8; ++r) {
      int m = r + 8 * grp;
      long long e = e0 + m;
      if (e < E_) wbuf[e * 64 + nc * 16 + ncol] = (_Float16)(d[r] + bbp);
    }
  }
}

// ============ 6. edge message scatter: per (channel, edge, copy) ============
__global__ void edge_message_kernel(const float* __restrict__ feats,
                                    const float* __restrict__ sh,
                                    const _Float16* __restrict__ wbuf,
                                    const int* __restrict__ edge,
                                    float* __restrict__ agg,
                                    int E_, int N_, int M_) {
  long long t = (long long)blockIdx.x * blockDim.x + threadIdx.x;
  long long total = (long long)E_ * M_ * CCH;
  if (t >= total) return;
  int c = (int)(t % CCH);
  long long em = t / CCH;
  int e = (int)(em % E_);
  int m = (int)(em / E_);
  float sh0 = sh[4 * e + 0];
  float s1x = sh[4 * e + 1], s1y = sh[4 * e + 2], s1z = sh[4 * e + 3];
  long long row = (long long)edge[e] + (long long)m * N_;
  long long col = (long long)edge[E_ + e] + (long long)m * N_;
  const float* fc = feats + col * FDIM;
  float sc = fc[c];
  float v0 = fc[16 + 3 * c], v1 = fc[16 + 3 * c + 1], v2 = fc[16 + 3 * c + 2];
  float w0 = (float)wbuf[(long long)e * 64 + c];
  float w1 = (float)wbuf[(long long)e * 64 + 16 + c];
  float w2 = (float)wbuf[(long long)e * 64 + 32 + c];
  float w3 = (float)wbuf[(long long)e * 64 + 48 + c];
  float dot = v0 * s1x + v1 * s1y + v2 * s1z;
  float ms  = w0 * sc * sh0 + w1 * dot;
  float mv0 = w2 * sc * s1x + w3 * v0 * sh0;
  float mv1 = w2 * sc * s1y + w3 * v1 * sh0;
  float mv2 = w2 * sc * s1z + w3 * v2 * sh0;
  float* ab = agg + row * FDIM;
  __hip_atomic_fetch_add(ab + c, ms, __ATOMIC_RELAXED, __HIP_MEMORY_SCOPE_AGENT);
  __hip_atomic_fetch_add(ab + 16 + 3 * c, mv0, __ATOMIC_RELAXED, __HIP_MEMORY_SCOPE_AGENT);
  __hip_atomic_fetch_add(ab + 16 + 3 * c + 1, mv1, __ATOMIC_RELAXED, __HIP_MEMORY_SCOPE_AGENT);
  __hip_atomic_fetch_add(ab + 16 + 3 * c + 2, mv2, __ATOMIC_RELAXED, __HIP_MEMORY_SCOPE_AGENT);
}

// ==== 7. node update via WMMA, SoA-staged LDS: out = feats + [agg_s@Ws|agg_v@Wv] ====
__global__ void node_update_kernel(const float* __restrict__ feats,
                                   const float* __restrict__ agg,
                                   const _Float16* __restrict__ Wst,
                                   const _Float16* __restrict__ Wvt,
                                   float* __restrict__ feats2, long long NM) {
  __shared__ _Float16 lds_s[8][16 * 32];      // agg scalar, K padded
  __shared__ _Float16 lds_v[8][3][16 * 32];   // agg vector comps, K padded
  int wave = threadIdx.x >> 5;
  int lane = threadIdx.x & 31;
  long long tile = (long long)blockIdx.x * 8 + wave;
  long long nm0 = tile * 16;
  if (nm0 < NM) {
    for (int idx = lane; idx < 16 * 128; idx += 32) {
      int m = idx >> 7;       // row 0..15
      int r = idx & 127;
      int which = r >> 5;     // 0=s, 1..3=v comp
      int k = r & 31;
      long long nm = nm0 + m;
      float val = 0.f;
      if (k < 16 && nm < NM)
        val = (which == 0) ? agg[nm * FDIM + k]
                           : agg[nm * FDIM + 16 + 3 * k + (which - 1)];
      _Float16 hv = (_Float16)val;
      if (which == 0) lds_s[wave][m * 32 + k] = hv;
      else            lds_v[wave][which - 1][m * 32 + k] = hv;
    }
  }
  asm volatile("s_wait_dscnt 0" ::: "memory");
  __builtin_amdgcn_wave_barrier();
  if (nm0 >= NM) return;
  int grp = lane >> 4;
  int mrow = lane & 15;
  int ncol = lane & 15;
  for (int X = 0; X < 4; ++X) {
    const _Float16* ab = (X == 0) ? &lds_s[wave][mrow * 32 + 16 * grp]
                                  : &lds_v[wave][X - 1][mrow * 32 + 16 * grp];
    afrag_u au;
    au.h[0] = *(const v8h*)(ab);
    au.h[1] = *(const v8h*)(ab + 8);
    v16h b = *(const v16h*)(((X == 0) ? Wst : Wvt) + ncol * 32 + grp * 16);
    v8f z = {};
    v8f d = __builtin_amdgcn_wmma_f32_16x16x32_f16(false, au.v, false, b, (short)0, z,
                                                   false, false);
    int dst = (X == 0) ? ncol : (16 + 3 * ncol + (X - 1));
#pragma unroll
    for (int r = 0; r < 8; ++r) {
      int m = r + 8 * grp;
      long long nm = nm0 + m;
      if (nm < NM) feats2[nm * FDIM + dst] = feats[nm * FDIM + dst] + d[r];
    }
  }
}

// ============ 8. batch segment bounds (batch_idx sorted) ============
__global__ void init_bounds_kernel(int* __restrict__ startb, int* __restrict__ endb) {
  int b = threadIdx.x;
  if (b < 8) { startb[b] = 0; endb[b] = 0; }
}
__global__ void find_bounds_kernel(const int* __restrict__ batch, int* __restrict__ startb,
                                   int* __restrict__ endb, int N_) {
  int i = blockIdx.x * blockDim.x + threadIdx.x;
  if (i >= N_) return;
  int b = batch[i];
  if (i == 0 || batch[i - 1] != b) startb[b] = i;
  if (i == N_ - 1 || batch[i + 1] != b) endb[b] = i + 1;
}

// ============ 9. segment-mean pooling over (batch, copy) ============
__global__ void pool_kernel(const float* __restrict__ feats2,
                            const int* __restrict__ startb,
                            const int* __restrict__ endb,
                            float* __restrict__ pooled, int N_, int M_) {
  int sm = blockIdx.x;        // b*M + m
  int b = sm / M_, m = sm % M_;
  int d = threadIdx.x;        // 0..63
  int s0 = startb[b], s1 = endb[b];
  float acc = 0.f;
  for (int i = s0; i < s1; ++i)
    acc += feats2[((long long)m * N_ + i) * FDIM + d];
  int cnt = s1 - s0;
  float div = (cnt > 0) ? (float)cnt : 1.f;
  pooled[sm * FDIM + d] = acc / div;
}

// ============ 10. final apply_D with D1(R_fwd) ============
__global__ void final_out_kernel(const float* __restrict__ pooled,
                                 const float* __restrict__ d1fwd,
                                 float* __restrict__ out, int SM) {
  int t = blockIdx.x * blockDim.x + threadIdx.x;
  if (t >= SM * CCH) return;
  int c = t % CCH;
  int sm = t / CCH;
  const float* D = d1fwd + sm * 9;
  const float* pb = pooled + sm * FDIM;
  float v0 = pb[16 + 3 * c], v1 = pb[16 + 3 * c + 1], v2 = pb[16 + 3 * c + 2];
  out[sm * FDIM + c] = pb[c];
  out[sm * FDIM + 16 + 3 * c]     = -(D[0] * v0 + D[1] * v1 + D[2] * v2);
  out[sm * FDIM + 16 + 3 * c + 1] = -(D[3] * v0 + D[4] * v1 + D[5] * v2);
  out[sm * FDIM + 16 + 3 * c + 2] = -(D[6] * v0 + D[7] * v1 + D[8] * v2);
}

extern "C" void kernel_launch(void* const* d_in, const int* in_sizes, int n_in,
                              void* d_out, int out_size, void* d_ws, size_t ws_size,
                              hipStream_t stream) {
  (void)n_in; (void)out_size; (void)ws_size;
  const float* x    = (const float*)d_in[0];
  const float* pos  = (const float*)d_in[1];
  const float* alg  = (const float*)d_in[2];
  const float* W1   = (const float*)d_in[3];
  const float* b1   = (const float*)d_in[4];
  const float* W2   = (const float*)d_in[5];
  const float* b2   = (const float*)d_in[6];
  const float* Wp   = (const float*)d_in[7];
  const float* bp   = (const float*)d_in[8];
  const float* Ws   = (const float*)d_in[9];
  const float* Wv   = (const float*)d_in[10];
  const int* edge   = (const int*)d_in[11];
  const int* batch  = (const int*)d_in[12];
  float* out        = (float*)d_out;

  const int N_ = in_sizes[12];
  const int E_ = in_sizes[11] / 2;
  const int M_ = 4, B_ = 8;
  const long long NM = (long long)N_ * M_;
  const int SM = B_ * M_;

  char* ws = (char*)d_ws;
  size_t off = 0;
  auto carve = [&](size_t bytes) -> void* {
    void* p = ws + off;
    off = (off + bytes + 255) & ~(size_t)255;
    return p;
  };
  float*    d1inv  = (float*)carve((size_t)SM * 9 * 4);
  float*    d1fwd  = (float*)carve((size_t)SM * 9 * 4);
  _Float16* wh     = (_Float16*)carve(4096 * 2);
  float*    elen   = (float*)carve((size_t)E_ * 4);
  float*    shb    = (float*)carve((size_t)E_ * 16);
  _Float16* wbuf   = (_Float16*)carve((size_t)E_ * 64 * 2);
  float*    feats  = (float*)carve((size_t)NM * FDIM * 4);
  float*    agg    = (float*)carve((size_t)NM * FDIM * 4);
  float*    feats2 = (float*)carve((size_t)NM * FDIM * 4);
  float*    pooled = (float*)carve((size_t)SM * FDIM * 4);
  int*      startb = (int*)carve(64);
  int*      endb   = (int*)carve(64);

  _Float16* W2t = wh;
  _Float16* Wpt = wh + 1024;
  _Float16* Wst = wh + 3072;
  _Float16* Wvt = wh + 3584;

  prep_poses_kernel<<<1, 32, 0, stream>>>(alg, d1inv, d1fwd, SM);
  convert_weights_kernel<<<(4096 + 255) / 256, 256, 0, stream>>>(W2, Wp, Ws, Wv, wh);
  hipMemsetAsync(agg, 0, (size_t)NM * FDIM * 4, stream);

  {
    long long total = NM * CCH;
    align_feats_kernel<<<(unsigned)((total + 255) / 256), 256, 0, stream>>>(
        x, batch, d1inv, feats, N_, M_);
  }
  edge_geom_kernel<<<(E_ + 255) / 256, 256, 0, stream>>>(pos, edge, elen, shb, E_);
  {
    int blocks = (E_ + 16 * 8 - 1) / (16 * 8);
    edge_mlp_kernel<<<blocks, 256, 0, stream>>>(elen, W1, b1, W2t, b2, Wpt, bp, wbuf, E_);
  }
  {
    long long total = (long long)E_ * M_ * CCH;
    edge_message_kernel<<<(unsigned)((total + 255) / 256), 256, 0, stream>>>(
        feats, shb, wbuf, edge, agg, E_, N_, M_);
  }
  {
    long long tiles = (NM + 15) / 16;
    int blocks = (int)((tiles + 7) / 8);
    node_update_kernel<<<blocks, 256, 0, stream>>>(feats, agg, Wst, Wvt, feats2, NM);
  }
  init_bounds_kernel<<<1, 8, 0, stream>>>(startb, endb);
  find_bounds_kernel<<<(N_ + 255) / 256, 256, 0, stream>>>(batch, startb, endb, N_);
  pool_kernel<<<SM, 64, 0, stream>>>(feats2, startb, endb, pooled, N_, M_);
  final_out_kernel<<<(SM * CCH + 255) / 256, 256, 0, stream>>>(pooled, d1fwd, out, SM);
}
